// FullMatchingModel_20572893348639
// MI455X (gfx1250) — compile-verified, matrix operand order checked by reading, not run on hardware
//
#include <hip/hip_runtime.h>
#include <cmath>
#include <cstddef>

// ---------------------------------------------------------------------------
// Types for CDNA5 WMMA (wave32, gfx1250)
// ---------------------------------------------------------------------------
typedef __bf16 bf16;
typedef __bf16 v16bf __attribute__((ext_vector_type(16)));
typedef float  v8f   __attribute__((ext_vector_type(8)));
typedef unsigned int u32x4 __attribute__((ext_vector_type(4)));

#define TPB 256          // threads per block (8 wave32 waves)
#define NB_RED 2048      // fixed reduction grid (deterministic partials)

// Problem constants (match setup_inputs)
#define N_P 2048
#define N_U 4096
#define D_X 768
#define D_Y 512
#define D_F 256
#define EPS_INV 20.0f    // 1/0.05 ; M = cos/eps  (dist = -cos)
#define SK_IT 10
#define TEMP_SAIL 10.0f

// Accumulator slots
#define S_MARG 0
#define S_SAIL 1
#define S_EXP  2
#define S_IMP  3
#define S_OT   4
#define S_N1   5
#define S_N2   6
#define S_DOT  7
#define S_GW1  8
#define S_GW2  9
#define S_GW3  10

// ---------------------------------------------------------------------------
// WMMA NT GEMM:  C(M,N) = alpha * A(M,K) * B(N,K)^T     (A,B bf16 row-major)
// Block = 8 waves in a 4x2 grid; each wave computes a 32x64 C tile as a
// 2x4 grid of 16x16 WMMA tiles (8 accumulators).  Per 32-K step: 6 fragment
// loads feed 8 v_wmma.  Block tile 128x128.
// Requires M%128==0, N%128==0, K%32==0 (true for every call here).
//
// Fragment layout (CDNA5 ISA 7.12.2, 16-bit A 16x32): lane = l + 16*h,
// lane holds row (l), K in [8h,8h+8) ++ [16+8h,16+8h+8).  B mirrors A with
// N in place of M, so row-major (N,K) data loads identically -> NT form.
//
// All six fragment base pointers (lane offset +8h folded in) are hoisted out
// of the K loop; the loop indexes them with "+ k0" only, which the unroller
// folds into the b128 instruction immediate offsets.  This keeps just 6 live
// 64-bit addresses and avoids the scratch spills seen with per-fragment
// address recomputation.
// ---------------------------------------------------------------------------
__device__ __forceinline__ v16bf load_frag_p(const bf16* __restrict__ p) {
  union { v16bf v; u32x4 q[2]; } f;
  f.q[0] = *(const u32x4*)(p);       // K = 8h .. 8h+7
  f.q[1] = *(const u32x4*)(p + 16);  // K = 16+8h .. 16+8h+7
  return f.v;
}

__global__ void k_gemm_nt(const bf16* __restrict__ A, const bf16* __restrict__ B,
                          float* __restrict__ C, int M, int N, int K, float alpha) {
  const int wave = threadIdx.x >> 5, lane = threadIdx.x & 31;
  const int l = lane & 15, h = lane >> 4;
  const int row0 = blockIdx.x * 128 + (wave & 3) * 32;   // 4 wave-rows of 32
  const int col0 = blockIdx.y * 128 + (wave >> 2) * 64;  // 2 wave-cols of 64
  if (row0 >= M || col0 >= N) return;  // wave-uniform; never taken for our dims

  v8f acc[2][4];
#pragma unroll
  for (int i = 0; i < 2; ++i)
#pragma unroll
    for (int j = 0; j < 4; ++j) acc[i][j] = (v8f){};

  // Hoisted fragment base pointers (include lane row and +8h K offset).
  const bf16* pa0 = A + (size_t)(row0 + l) * K + 8 * h;
  const bf16* pa1 = pa0 + (size_t)16 * K;
  const bf16* pb0 = B + (size_t)(col0 + l) * K + 8 * h;
  const bf16* pb1 = pb0 + (size_t)16 * K;
  const bf16* pb2 = pb0 + (size_t)32 * K;
  const bf16* pb3 = pb0 + (size_t)48 * K;

  for (int k0 = 0; k0 < K; k0 += 32) {
    // speculative prefetch of the next K panel (global_prefetch_b8);
    // out-of-range addresses at the tail are silently dropped.
    __builtin_prefetch(pa0 + k0 + 256, 0, 1);
    __builtin_prefetch(pb0 + k0 + 256, 0, 1);

    v16bf a0 = load_frag_p(pa0 + k0);
    v16bf a1 = load_frag_p(pa1 + k0);
    v16bf b0 = load_frag_p(pb0 + k0);
    v16bf b1 = load_frag_p(pb1 + k0);
    v16bf b2 = load_frag_p(pb2 + k0);
    v16bf b3 = load_frag_p(pb3 + k0);

    acc[0][0] = __builtin_amdgcn_wmma_f32_16x16x32_bf16(false, a0, false, b0, (short)0, acc[0][0], false, false);
    acc[0][1] = __builtin_amdgcn_wmma_f32_16x16x32_bf16(false, a0, false, b1, (short)0, acc[0][1], false, false);
    acc[0][2] = __builtin_amdgcn_wmma_f32_16x16x32_bf16(false, a0, false, b2, (short)0, acc[0][2], false, false);
    acc[0][3] = __builtin_amdgcn_wmma_f32_16x16x32_bf16(false, a0, false, b3, (short)0, acc[0][3], false, false);
    acc[1][0] = __builtin_amdgcn_wmma_f32_16x16x32_bf16(false, a1, false, b0, (short)0, acc[1][0], false, false);
    acc[1][1] = __builtin_amdgcn_wmma_f32_16x16x32_bf16(false, a1, false, b1, (short)0, acc[1][1], false, false);
    acc[1][2] = __builtin_amdgcn_wmma_f32_16x16x32_bf16(false, a1, false, b2, (short)0, acc[1][2], false, false);
    acc[1][3] = __builtin_amdgcn_wmma_f32_16x16x32_bf16(false, a1, false, b3, (short)0, acc[1][3], false, false);
  }

  // C/D layout: VGPR r -> M = r + 8h ; lane l -> N = l
#pragma unroll
  for (int i = 0; i < 2; ++i)
#pragma unroll
    for (int j = 0; j < 4; ++j) {
      float* Cp = C + (size_t)(row0 + 16 * i) * N + col0 + 16 * j;
#pragma unroll
      for (int r = 0; r < 8; ++r)
        Cp[(size_t)(r + 8 * h) * N + l] = alpha * acc[i][j][r];
    }
}

// ---------------------------------------------------------------------------
// Data movement / conversion kernels
// ---------------------------------------------------------------------------
__global__ void k_fill(float* p, float v, int n) {
  int i = blockIdx.x * blockDim.x + threadIdx.x;
  if (i < n) p[i] = v;
}

__global__ void k_convert_bf16(const float* __restrict__ s, bf16* __restrict__ d, size_t n) {
  size_t stride = (size_t)gridDim.x * blockDim.x;
  for (size_t i = blockIdx.x * (size_t)blockDim.x + threadIdx.x; i < n; i += stride)
    d[i] = (bf16)s[i];
}

template <typename T>
__global__ void k_transpose_to_bf16(const T* __restrict__ src, bf16* __restrict__ dst,
                                    int R, int C) {  // src RxC -> dst CxR
  __shared__ float tile[32][33];
  int c0 = blockIdx.x * 32, r0 = blockIdx.y * 32;
  for (int i = threadIdx.y; i < 32; i += 8) {
    int r = r0 + i, c = c0 + threadIdx.x;
    tile[i][threadIdx.x] = (r < R && c < C) ? (float)src[(size_t)r * C + c] : 0.f;
  }
  __syncthreads();
  for (int i = threadIdx.y; i < 32; i += 8) {
    int r = c0 + i, c = r0 + threadIdx.x;
    if (r < C && c < R) dst[(size_t)r * R + c] = (bf16)tile[threadIdx.x][i];
  }
}

__global__ void k_rownorm_bf16(const float* __restrict__ src, bf16* __restrict__ dst, int C) {
  int row = blockIdx.x;
  const float* p = src + (size_t)row * C;
  __shared__ float red[TPB];
  float s = 0.f;
  for (int j = threadIdx.x; j < C; j += TPB) { float x = p[j]; s += x * x; }
  red[threadIdx.x] = s; __syncthreads();
  for (int k = TPB / 2; k > 0; k >>= 1) {
    if (threadIdx.x < k) red[threadIdx.x] += red[threadIdx.x + k];
    __syncthreads();
  }
  float inv = 1.f / fmaxf(sqrtf(red[0]), 1e-8f);
  for (int j = threadIdx.x; j < C; j += TPB)
    dst[(size_t)row * C + j] = (bf16)(p[j] * inv);
}

// n_i = clip(sqrt(sum X_i * (Sxx X)_i), 1e-8);  dst = X / n  (bf16)
__global__ void k_quadnorm_bf16(const float* __restrict__ X, const float* __restrict__ T0,
                                bf16* __restrict__ dst, int C) {
  int row = blockIdx.x;
  const float* px = X + (size_t)row * C;
  const float* pt = T0 + (size_t)row * C;
  __shared__ float red[TPB];
  float s = 0.f;
  for (int j = threadIdx.x; j < C; j += TPB) s += px[j] * pt[j];
  red[threadIdx.x] = s; __syncthreads();
  for (int k = TPB / 2; k > 0; k >>= 1) {
    if (threadIdx.x < k) red[threadIdx.x] += red[threadIdx.x + k];
    __syncthreads();
  }
  float inv = 1.f / fmaxf(sqrtf(fmaxf(red[0], 0.f)), 1e-8f);
  for (int j = threadIdx.x; j < C; j += TPB)
    dst[(size_t)row * C + j] = (bf16)(px[j] * inv);
}

// ---------------------------------------------------------------------------
// Log-domain Sinkhorn (M = scale*S implicit; never materialized)
// ---------------------------------------------------------------------------
__device__ __forceinline__ void lse_combine(float& m, float& s, float m2, float s2) {
  if (m2 == -__builtin_inff()) return;
  if (m == -__builtin_inff()) { m = m2; s = s2; return; }
  if (m2 > m) { s = s * expf(m - m2) + s2; m = m2; }
  else         s = s + s2 * expf(m2 - m);
}

__global__ void k_lse_rows(const float* __restrict__ S, float scale,
                           const float* __restrict__ v, float* __restrict__ u,
                           int ny, float log_a) {
  int i = blockIdx.x;
  const float* row = S + (size_t)i * ny;
  float m = -__builtin_inff(), s = 0.f;
  for (int j = threadIdx.x; j < ny; j += TPB) {
    float x = scale * row[j] + v[j];
    if (x > m) { s = s * expf(m - x) + 1.f; m = x; } else s += expf(x - m);
  }
  __shared__ float sm[TPB], ss[TPB];
  sm[threadIdx.x] = m; ss[threadIdx.x] = s; __syncthreads();
  for (int k = TPB / 2; k > 0; k >>= 1) {
    if (threadIdx.x < k) {
      float mm = sm[threadIdx.x], s1 = ss[threadIdx.x];
      lse_combine(mm, s1, sm[threadIdx.x + k], ss[threadIdx.x + k]);
      sm[threadIdx.x] = mm; ss[threadIdx.x] = s1;
    }
    __syncthreads();
  }
  if (threadIdx.x == 0) u[i] = log_a - (sm[0] + logf(ss[0]));
}

__global__ void k_lse_cols(const float* __restrict__ S, float scale,
                           const float* __restrict__ u, float* __restrict__ v,
                           int nx, int ny, float log_b) {
  int c = blockIdx.x * blockDim.x + threadIdx.x;
  if (c >= ny) return;
  float m = -__builtin_inff(), s = 0.f;
  for (int i = 0; i < nx; ++i) {
    float x = scale * S[(size_t)i * ny + c] + u[i];
    if (x > m) { s = s * expf(m - x) + 1.f; m = x; } else s += expf(x - m);
  }
  v[c] = log_b - (m + logf(s));
}

__global__ void k_plan_mat(const float* __restrict__ S, float scale,
                           const float* __restrict__ u, const float* __restrict__ v,
                           float* __restrict__ P, int nx, int ny) {
  size_t total = (size_t)nx * ny, stride = (size_t)gridDim.x * blockDim.x;
  for (size_t idx = blockIdx.x * (size_t)blockDim.x + threadIdx.x; idx < total; idx += stride) {
    int i = (int)(idx / ny), j = (int)(idx - (size_t)i * ny);
    P[idx] = expf(scale * S[idx] + u[i] + v[j]);
  }
}

__global__ void k_plan_rowsum(const float* __restrict__ S, float scale,
                              const float* __restrict__ u, const float* __restrict__ v,
                              float* __restrict__ rs, int ny) {
  int i = blockIdx.x;
  const float* row = S + (size_t)i * ny;
  float ui = u[i];
  __shared__ float red[TPB];
  float s = 0.f;
  for (int j = threadIdx.x; j < ny; j += TPB) s += expf(scale * row[j] + ui + v[j]);
  red[threadIdx.x] = s; __syncthreads();
  for (int k = TPB / 2; k > 0; k >>= 1) {
    if (threadIdx.x < k) red[threadIdx.x] += red[threadIdx.x + k];
    __syncthreads();
  }
  if (threadIdx.x == 0) rs[i] = red[0];
}

__global__ void k_plan_colsum(const float* __restrict__ S, float scale,
                              const float* __restrict__ u, const float* __restrict__ v,
                              float* __restrict__ cs, int nx, int ny) {
  int c = blockIdx.x * blockDim.x + threadIdx.x;
  if (c >= ny) return;
  float vc = v[c], s = 0.f;
  for (int i = 0; i < nx; ++i) s += expf(scale * S[(size_t)i * ny + c] + u[i] + vc);
  cs[c] = s;
}

// ---------------------------------------------------------------------------
// Loss reductions (deterministic: fixed-grid partials + single-block sum)
// ---------------------------------------------------------------------------
__device__ __forceinline__ float block_reduce(float val) {
  __shared__ float red[TPB];
  red[threadIdx.x] = val; __syncthreads();
  for (int k = TPB / 2; k > 0; k >>= 1) {
    if (threadIdx.x < k) red[threadIdx.x] += red[threadIdx.x + k];
    __syncthreads();
  }
  return red[0];
}

__global__ void k_marg(const float* __restrict__ r, int n, float* acc) {  // 1 block
  float inv = 1.f / (float)n, s = 0.f;
  for (int i = threadIdx.x; i < n; i += TPB) { float d = r[i] - inv; s += d * d; }
  float tot = block_reduce(s);
  if (threadIdx.x == 0) acc[S_MARG] += tot;
}

__global__ void k_sail(const float* __restrict__ cosP, int N, float* __restrict__ part) {
  size_t total = (size_t)N * N, stride = (size_t)gridDim.x * blockDim.x;
  float s = 0.f;
  for (size_t idx = blockIdx.x * (size_t)blockDim.x + threadIdx.x; idx < total; idx += stride) {
    int i = (int)(idx / N), j = (int)(idx - (size_t)i * N);
    float t = (i == j) ? 1.f : -1.f;
    float z = TEMP_SAIL * cosP[idx] * t;
    s += fminf(z, 0.f) - log1pf(expf(-fabsf(z)));  // log_sigmoid(z)
  }
  float tot = block_reduce(s);
  if (threadIdx.x == 0) part[blockIdx.x] = tot;
}

__global__ void k_pairs_diag(const float* __restrict__ cosP, const float* __restrict__ u,
                             const float* __restrict__ v, int N, float* acc) {  // 1 block
  float e = 0.f, m = 0.f, lgN = logf((float)N);
  for (int i = threadIdx.x; i < N; i += TPB) {
    float c = cosP[(size_t)i * N + i];
    e += (1.f - c) * 0.5f;
    m += c * EPS_INV + u[i] + v[i] + lgN;  // diag(logp) + log N
  }
  __shared__ float r2[TPB];
  float te = block_reduce(e);
  r2[threadIdx.x] = m; __syncthreads();
  for (int k = TPB / 2; k > 0; k >>= 1) {
    if (threadIdx.x < k) r2[threadIdx.x] += r2[threadIdx.x + k];
    __syncthreads();
  }
  if (threadIdx.x == 0) { acc[S_EXP] += te / N; acc[S_IMP] += -r2[0] / N; }
}

__global__ void k_ot(const float* __restrict__ Sa, const float* __restrict__ ua,
                     const float* __restrict__ va, const float* __restrict__ Su,
                     const float* __restrict__ uu, const float* __restrict__ vu,
                     int ny, size_t total, float* __restrict__ part) {
  size_t stride = (size_t)gridDim.x * blockDim.x;
  float s = 0.f;
  for (size_t idx = blockIdx.x * (size_t)blockDim.x + threadIdx.x; idx < total; idx += stride) {
    int i = (int)(idx / ny), j = (int)(idx - (size_t)i * ny);
    float la = EPS_INV * Sa[idx] + ua[i] + va[j];
    float lu = EPS_INV * Su[idx] + uu[i] + vu[j];
    s += expf(la) * (la - lu);
  }
  float tot = block_reduce(s);
  if (threadIdx.x == 0) part[blockIdx.x] = tot;
}

__global__ void k_dot_reduce(const float* __restrict__ A, const float* __restrict__ B,
                             size_t n, float* __restrict__ part) {
  size_t stride = (size_t)gridDim.x * blockDim.x;
  float s = 0.f;
  for (size_t i = blockIdx.x * (size_t)blockDim.x + threadIdx.x; i < n; i += stride)
    s += A[i] * B[i];
  float tot = block_reduce(s);
  if (threadIdx.x == 0) part[blockIdx.x] = tot;
}

// sum_ij w_i * K_ij^2 * w_j
__global__ void k_quad_reduce(const float* __restrict__ K, const float* __restrict__ wv,
                              int ny, size_t total, float* __restrict__ part) {
  size_t stride = (size_t)gridDim.x * blockDim.x;
  float s = 0.f;
  for (size_t idx = blockIdx.x * (size_t)blockDim.x + threadIdx.x; idx < total; idx += stride) {
    int i = (int)(idx / ny), j = (int)(idx - (size_t)i * ny);
    float x = K[idx];
    s += wv[i] * x * x * wv[j];
  }
  float tot = block_reduce(s);
  if (threadIdx.x == 0) part[blockIdx.x] = tot;
}

__global__ void k_sum_partials(const float* __restrict__ part, int n,
                               float* acc, int slot, float coef) {  // 1 block
  float s = 0.f;
  for (int i = threadIdx.x; i < n; i += TPB) s += part[i];
  float tot = block_reduce(s);
  if (threadIdx.x == 0) acc[slot] += coef * tot;
}

__global__ void k_combine(const float* __restrict__ acc, float* __restrict__ out) {
  if (threadIdx.x == 0) {
    float cste = (float)N_U * (float)N_U;
    float L_div = (acc[S_N1] + acc[S_N2] - 2.f * acc[S_DOT]) / cste;
    float L_gw  = acc[S_GW1] + acc[S_GW2] - 2.f * acc[S_GW3];
    out[0] = acc[S_MARG] + acc[S_SAIL] + acc[S_EXP] + acc[S_IMP] + acc[S_OT] + L_div + L_gw;
  }
}

// ---------------------------------------------------------------------------
// Host orchestration
// ---------------------------------------------------------------------------
extern "C" void kernel_launch(void* const* d_in, const int* in_sizes, int n_in,
                              void* d_out, int out_size, void* d_ws, size_t ws_size,
                              hipStream_t stream) {
  (void)in_sizes; (void)n_in; (void)out_size; (void)ws_size;
  const float* fXp = (const float*)d_in[0];
  const float* fYp = (const float*)d_in[1];
  const float* X   = (const float*)d_in[2];
  const float* Y   = (const float*)d_in[3];
  const float* fX  = (const float*)d_in[4];
  const float* fY  = (const float*)d_in[5];
  const float* Xaf = (const float*)d_in[6];
  const float* Yaf = (const float*)d_in[7];
  float* out = (float*)d_out;

  // ---- workspace layout (running 256B-aligned allocator, ~520 MB) ----
  char* w = (char*)d_ws;
  auto af = [&](size_t n) { float* p = (float*)w; w += ((n * 4 + 255) & ~(size_t)255); return p; };
  auto ab = [&](size_t n) { bf16*  p = (bf16*)w;  w += ((n * 2 + 255) & ~(size_t)255); return p; };

  float* acc   = af(64);
  float* part  = af(NB_RED);
  float* S_p   = af((size_t)N_P * N_P);
  float* u_p   = af(N_P); float* v_p = af(N_P);
  float* rs_p  = af(N_P); float* cs_p = af(N_P);
  float* S_u   = af((size_t)N_U * N_U);
  float* u_u   = af(N_U); float* v_u = af(N_U);
  float* S_a   = af((size_t)N_U * N_U);
  float* u_a   = af(N_U); float* v_a = af(N_U);
  float* P_u   = af((size_t)N_U * N_U);
  float* a_u   = af(N_U); float* b_u = af(N_U);
  bf16*  Put   = ab((size_t)N_U * N_U);
  bf16*  Kxb   = ab((size_t)N_U * N_U);
  bf16*  Kyb   = ab((size_t)N_U * N_U);
  bf16*  T2b   = ab((size_t)N_U * N_U);
  float* Gbig  = af((size_t)N_U * N_U);          // reused: Kx -> Ky -> Kx@P -> W
  bf16*  fXpn  = ab((size_t)N_P * D_F);
  bf16*  fYpn  = ab((size_t)N_P * D_F);
  bf16*  fXnb  = ab((size_t)N_U * D_F);
  bf16*  fYnb  = ab((size_t)N_U * D_F);
  bf16*  fXnt  = ab((size_t)D_F * N_U);
  bf16*  fYnt  = ab((size_t)D_F * N_U);
  bf16*  Xab   = ab((size_t)N_U * D_X);
  bf16*  Yab   = ab((size_t)N_U * D_Y);
  bf16*  Xat   = ab((size_t)D_X * N_U);
  bf16*  Yat   = ab((size_t)D_Y * N_U);
  bf16*  Xb    = ab((size_t)N_U * D_X);
  bf16*  Yb    = ab((size_t)N_U * D_Y);
  bf16*  Xub   = ab((size_t)N_U * D_X);
  bf16*  Yub   = ab((size_t)N_U * D_Y);
  bf16*  Xnb   = ab((size_t)N_U * D_X);
  bf16*  Ynb   = ab((size_t)N_U * D_Y);
  bf16*  Xnt   = ab((size_t)D_X * N_U);
  bf16*  Ynt   = ab((size_t)D_Y * N_U);
  float* Sxx   = af((size_t)D_X * D_X);
  float* Syy   = af((size_t)D_Y * D_Y);
  float* Sxy   = af((size_t)D_X * D_Y);
  bf16*  Sxxb  = ab((size_t)D_X * D_X);
  bf16*  Syyb  = ab((size_t)D_Y * D_Y);
  bf16*  Sxytb = ab((size_t)D_Y * D_X);
  float* T0    = af((size_t)N_U * D_X);          // reused for X and Y quad-forms
  float* T1    = af((size_t)N_U * D_Y);
  bf16*  T1b   = ab((size_t)N_U * D_Y);
  float* Gxf   = af((size_t)D_X * D_X);  bf16* Gxb  = ab((size_t)D_X * D_X);
  float* Gyf   = af((size_t)D_Y * D_Y);  bf16* Gyb  = ab((size_t)D_Y * D_Y);
  float* U1f   = af((size_t)D_X * D_Y);  bf16* U1b  = ab((size_t)D_X * D_Y);
  float* U2f   = af((size_t)D_X * D_Y);
  float* P1f   = af((size_t)D_F * D_F);
  float* P2f   = af((size_t)D_F * D_F);
  float* A1f   = af((size_t)D_X * D_F);  bf16* A1b  = ab((size_t)D_X * D_F);
  float* B1f   = af((size_t)D_Y * D_F);  bf16* B1tb = ab((size_t)D_Y * D_F);
  float* Vf    = af((size_t)D_X * D_Y);

  // ---- helpers ----
  auto gemm = [&](const bf16* A, const bf16* B, float* C, int M, int N, int K, float alpha) {
    k_gemm_nt<<<dim3(M / 128, N / 128), TPB, 0, stream>>>(A, B, C, M, N, K, alpha);
  };
  auto cvt = [&](const float* s, bf16* d, size_t n) {
    k_convert_bf16<<<NB_RED, TPB, 0, stream>>>(s, d, n);
  };
  auto tr_f = [&](const float* s, bf16* d, int R, int C) {
    k_transpose_to_bf16<float><<<dim3((C + 31) / 32, (R + 31) / 32), dim3(32, 8), 0, stream>>>(s, d, R, C);
  };
  auto tr_b = [&](const bf16* s, bf16* d, int R, int C) {
    k_transpose_to_bf16<bf16><<<dim3((C + 31) / 32, (R + 31) / 32), dim3(32, 8), 0, stream>>>(s, d, R, C);
  };
  auto sinkhorn = [&](const float* S, float* u, float* v, int nx, int ny) {
    k_fill<<<(nx + TPB - 1) / TPB, TPB, 0, stream>>>(u, 0.f, nx);
    k_fill<<<(ny + TPB - 1) / TPB, TPB, 0, stream>>>(v, 0.f, ny);
    float la = -logf((float)nx), lb = -logf((float)ny);
    for (int it = 0; it < SK_IT; ++it) {
      k_lse_rows<<<nx, TPB, 0, stream>>>(S, EPS_INV, v, u, ny, la);
      k_lse_cols<<<(ny + TPB - 1) / TPB, TPB, 0, stream>>>(S, EPS_INV, u, v, nx, ny, lb);
    }
  };

  k_fill<<<1, 64, 0, stream>>>(acc, 0.f, 64);

  // ---- normalizations & bf16 copies ----
  k_rownorm_bf16<<<N_P, TPB, 0, stream>>>(fXp, fXpn, D_F);
  k_rownorm_bf16<<<N_P, TPB, 0, stream>>>(fYp, fYpn, D_F);
  k_rownorm_bf16<<<N_U, TPB, 0, stream>>>(fX, fXnb, D_F);
  k_rownorm_bf16<<<N_U, TPB, 0, stream>>>(fY, fYnb, D_F);
  k_rownorm_bf16<<<N_U, TPB, 0, stream>>>(Xaf, Xab, D_X);
  k_rownorm_bf16<<<N_U, TPB, 0, stream>>>(Yaf, Yab, D_Y);
  k_rownorm_bf16<<<N_U, TPB, 0, stream>>>(X, Xub, D_X);   // Xu = rownorm(X)
  k_rownorm_bf16<<<N_U, TPB, 0, stream>>>(Y, Yub, D_Y);
  cvt(X, Xb, (size_t)N_U * D_X);
  cvt(Y, Yb, (size_t)N_U * D_Y);
  tr_b(Xab, Xat, N_U, D_X);
  tr_b(Yab, Yat, N_U, D_Y);
  tr_b(fXnb, fXnt, N_U, D_F);
  tr_b(fYnb, fYnt, N_U, D_F);

  // ---- cosine sims and anchor covariances ----
  gemm(fXpn, fYpn, S_p, N_P, N_P, D_F, 1.f);                    // cos_p
  gemm(fXnb, fYnb, S_u, N_U, N_U, D_F, 1.f);                    // cos_u
  gemm(Xat, Xat, Sxx, D_X, D_X, N_U, 1.f / (float)N_U);         // Sxx
  gemm(Yat, Yat, Syy, D_Y, D_Y, N_U, 1.f / (float)N_U);         // Syy
  gemm(Xat, Yat, Sxy, D_X, D_Y, N_U, 1.f / (float)N_U);         // Sxy
  cvt(Sxx, Sxxb, (size_t)D_X * D_X);
  cvt(Syy, Syyb, (size_t)D_Y * D_Y);
  tr_f(Sxy, Sxytb, D_X, D_Y);                                   // Sxy^T (bf16)

  // ---- sinkhorn on pairs + pair losses ----
  sinkhorn(S_p, u_p, v_p, N_P, N_P);
  k_plan_rowsum<<<N_P, TPB, 0, stream>>>(S_p, EPS_INV, u_p, v_p, rs_p, N_P);
  k_plan_colsum<<<(N_P + TPB - 1) / TPB, TPB, 0, stream>>>(S_p, EPS_INV, u_p, v_p, cs_p, N_P, N_P);
  k_marg<<<1, TPB, 0, stream>>>(rs_p, N_P, acc);
  k_marg<<<1, TPB, 0, stream>>>(cs_p, N_P, acc);
  k_sail<<<NB_RED, TPB, 0, stream>>>(S_p, N_P, part);
  k_sum_partials<<<1, TPB, 0, stream>>>(part, NB_RED, acc, S_SAIL, -1.f / ((float)N_P * (float)N_P));
  k_pairs_diag<<<1, TPB, 0, stream>>>(S_p, u_p, v_p, N_P, acc);

  // ---- sinkhorn on unpaired + plan_u ----
  sinkhorn(S_u, u_u, v_u, N_U, N_U);
  k_plan_mat<<<NB_RED, TPB, 0, stream>>>(S_u, EPS_INV, u_u, v_u, P_u, N_U, N_U);
  k_plan_rowsum<<<N_U, TPB, 0, stream>>>(S_u, EPS_INV, u_u, v_u, a_u, N_U);
  k_plan_colsum<<<(N_U + TPB - 1) / TPB, TPB, 0, stream>>>(S_u, EPS_INV, u_u, v_u, b_u, N_U, N_U);
  k_marg<<<1, TPB, 0, stream>>>(a_u, N_U, acc);
  k_marg<<<1, TPB, 0, stream>>>(b_u, N_U, acc);
  tr_f(P_u, Put, N_U, N_U);                                     // plan_u^T (bf16)

  // ---- anchor-metric normalization: Xn, Yn ----
  gemm(Xb, Sxxb, T0, N_U, D_X, D_X, 1.f);                       // X @ Sxx (sym)
  k_quadnorm_bf16<<<N_U, TPB, 0, stream>>>(X, T0, Xnb, D_X);
  gemm(Yb, Syyb, T0, N_U, D_Y, D_Y, 1.f);                       // Y @ Syy (sym)
  k_quadnorm_bf16<<<N_U, TPB, 0, stream>>>(Y, T0, Ynb, D_Y);
  tr_b(Xnb, Xnt, N_U, D_X);
  tr_b(Ynb, Ynt, N_U, D_Y);

  // ---- anchor-space cost + sinkhorn + L_ot ----
  gemm(Xnb, Sxytb, T1, N_U, D_Y, D_X, 1.f);                     // Xn @ Sxy
  cvt(T1, T1b, (size_t)N_U * D_Y);
  gemm(T1b, Ynb, S_a, N_U, N_U, D_Y, 1.f);                      // (Xn Sxy) Yn^T
  sinkhorn(S_a, u_a, v_a, N_U, N_U);
  k_ot<<<NB_RED, TPB, 0, stream>>>(S_a, u_a, v_a, S_u, u_u, v_u, N_U, (size_t)N_U * N_U, part);
  k_sum_partials<<<1, TPB, 0, stream>>>(part, NB_RED, acc, S_OT, 1.f);

  // ---- L_div ----
  gemm(Xnt, Xnt, Gxf, D_X, D_X, N_U, 1.f);  cvt(Gxf, Gxb, (size_t)D_X * D_X);   // Xn^T Xn
  gemm(Ynt, Ynt, Gyf, D_Y, D_Y, N_U, 1.f);  cvt(Gyf, Gyb, (size_t)D_Y * D_Y);   // Yn^T Yn
  gemm(Gxb, Sxytb, U1f, D_X, D_Y, D_X, 1.f);  cvt(U1f, U1b, (size_t)D_X * D_Y); // Gx Sxy
  gemm(U1b, Gyb, U2f, D_X, D_Y, D_Y, 1.f);                                      // (Gx Sxy) Gy
  k_dot_reduce<<<NB_RED, TPB, 0, stream>>>(U2f, Sxy, (size_t)D_X * D_Y, part);
  k_sum_partials<<<1, TPB, 0, stream>>>(part, NB_RED, acc, S_N1, 1.f);
  gemm(fXnt, fXnt, P1f, D_F, D_F, N_U, 1.f);
  gemm(fYnt, fYnt, P2f, D_F, D_F, N_U, 1.f);
  k_dot_reduce<<<NB_RED, TPB, 0, stream>>>(P1f, P2f, (size_t)D_F * D_F, part);
  k_sum_partials<<<1, TPB, 0, stream>>>(part, NB_RED, acc, S_N2, 1.f);
  gemm(Xnt, fXnt, A1f, D_X, D_F, N_U, 1.f);  cvt(A1f, A1b, (size_t)D_X * D_F);  // Xn^T fXn
  gemm(Ynt, fYnt, B1f, D_Y, D_F, N_U, 1.f);  cvt(B1f, B1tb, (size_t)D_Y * D_F); // Yn^T fYn
  gemm(A1b, B1tb, Vf, D_X, D_Y, D_F, 1.f);                                      // (Xn^T fXn)(fYn^T Yn)
  k_dot_reduce<<<NB_RED, TPB, 0, stream>>>(Vf, Sxy, (size_t)D_X * D_Y, part);
  k_sum_partials<<<1, TPB, 0, stream>>>(part, NB_RED, acc, S_DOT, 1.f);

  // ---- L_gw ----
  gemm(Xub, Xub, Gbig, N_U, N_U, D_X, 1.f);                     // Kx
  k_quad_reduce<<<NB_RED, TPB, 0, stream>>>(Gbig, a_u, N_U, (size_t)N_U * N_U, part);
  k_sum_partials<<<1, TPB, 0, stream>>>(part, NB_RED, acc, S_GW1, 1.f);
  cvt(Gbig, Kxb, (size_t)N_U * N_U);
  gemm(Yub, Yub, Gbig, N_U, N_U, D_Y, 1.f);                     // Ky
  k_quad_reduce<<<NB_RED, TPB, 0, stream>>>(Gbig, b_u, N_U, (size_t)N_U * N_U, part);
  k_sum_partials<<<1, TPB, 0, stream>>>(part, NB_RED, acc, S_GW2, 1.f);
  cvt(Gbig, Kyb, (size_t)N_U * N_U);
  gemm(Kxb, Put, Gbig, N_U, N_U, N_U, 1.f);                     // Kx @ plan_u
  cvt(Gbig, T2b, (size_t)N_U * N_U);
  gemm(T2b, Kyb, Gbig, N_U, N_U, N_U, 1.f);                     // (Kx plan_u) Ky
  k_dot_reduce<<<NB_RED, TPB, 0, stream>>>(Gbig, P_u, (size_t)N_U * N_U, part);
  k_sum_partials<<<1, TPB, 0, stream>>>(part, NB_RED, acc, S_GW3, 1.f);

  // ---- final scalar ----
  k_combine<<<1, 32, 0, stream>>>(acc, out);
}
